// LRU_2817498546780
// MI455X (gfx1250) — matrix operand order repaired and checked by reading, not compiled, not run on hardware
//
#include <hip/hip_runtime.h>
#include <hip/hip_bf16.h>
#include <stddef.h>

// ---------------------------------------------------------------------------
// LRU forward for MI455X (gfx1250, wave32, WMMA).
//   N = H = 512, L = 16384
// Pipeline:
//   prep_lambda : lambda_eff = exp(-exp(nu)+i*exp(theta)) * (state_re+i*state_im)
//   prep_B      : B' = (B_re+i B_im)*exp(i*gamma)  -> bf16 (N,H) row-major
//   prep_C      : C_re -> bf16, -C_im -> bf16      -> (H,N) row-major
//   conv_x      : x -> bf16 (L,H)
//   gemm1       : Bu_re/Bu_im (L,N) f32 = x_bf @ B'^T   [v_wmma_f32_16x16x32_bf16]
//   scan_local  : per 64-row chunk local recurrence (in-place on Bu)
//   scan_carry  : serial scan of 256 chunk aggregates (lambda^64 via squaring)
//   apply_carry : h = local + lambda^{t+1}*carry  -> bf16 h_re/h_im (L,N)
//   gemm2       : y = h_re@C_re^T + h_im@(-C_im)^T + D*x   [WMMA, fused epilogue]
// ---------------------------------------------------------------------------

#define LRU_N 512
#define LRU_H 512
#define LRU_L 16384
#define LRU_CHUNK 64
#define LRU_NCHUNK (LRU_L / LRU_CHUNK)   // 256

typedef __attribute__((ext_vector_type(16))) __bf16 v16bf;
typedef __attribute__((ext_vector_type(8)))  __bf16 v8bf;
typedef __attribute__((ext_vector_type(8)))  float  v8f;

static __device__ __forceinline__ v16bf cat16(v8bf lo, v8bf hi) {
    return __builtin_shufflevector(lo, hi, 0, 1, 2, 3, 4, 5, 6, 7,
                                           8, 9, 10, 11, 12, 13, 14, 15);
}

// A fragment (16x32 bf16, M x K): lane = 16*kh + m; runs K = kh*8..+7 and kh*8+16..+7
static __device__ __forceinline__ v16bf load_fragA(const __bf16* base, int row_stride,
                                                   int row, int k0, int lane) {
    const int m  = lane & 15;
    const int kh = lane >> 4;
    const __bf16* p = base + (size_t)(row + m) * row_stride + k0 + kh * 8;
    v8bf lo = *(const v8bf*)(p);
    v8bf hi = *(const v8bf*)(p + 16);
    return cat16(lo, hi);
}

// B fragment (32x16 bf16, K x N), source row-major [ncol][k]:
// lane = 16*kh + n; runs K = kh*8..+7 and kh*8+16..+7
static __device__ __forceinline__ v16bf load_fragB(const __bf16* base, int row_stride,
                                                   int col0, int k0, int lane) {
    const int n  = lane & 15;
    const int kh = lane >> 4;
    const __bf16* p = base + (size_t)(col0 + n) * row_stride + k0 + kh * 8;
    v8bf lo = *(const v8bf*)(p);
    v8bf hi = *(const v8bf*)(p + 16);
    return cat16(lo, hi);
}

// ------------------------------- prep kernels ------------------------------

__global__ void lru_prep_lambda(const float* __restrict__ nu_log,
                                const float* __restrict__ theta_log,
                                const float* __restrict__ state_re,
                                const float* __restrict__ state_im,
                                float* __restrict__ lam_re,
                                float* __restrict__ lam_im) {
    int n = threadIdx.x;                         // one block of 512 threads
    float nu = __expf(nu_log[n]);
    float th = __expf(theta_log[n]);
    float r  = __expf(-nu);
    float lre = r * __cosf(th);
    float lim = r * __sinf(th);
    float sre = state_re[n], sim = state_im[n];
    lam_re[n] = lre * sre - lim * sim;
    lam_im[n] = lre * sim + lim * sre;
}

__global__ void lru_prep_B(const float* __restrict__ B_re,
                           const float* __restrict__ B_im,
                           const float* __restrict__ gamma_log,
                           __bf16* __restrict__ Bp_re,
                           __bf16* __restrict__ Bp_im) {
    int idx = blockIdx.x * blockDim.x + threadIdx.x;   // N*H
    int n = idx / LRU_H;
    float g  = gamma_log[n];
    float cg = __cosf(g), sg = __sinf(g);
    float br = B_re[idx], bi = B_im[idx];
    Bp_re[idx] = (__bf16)(br * cg - bi * sg);
    Bp_im[idx] = (__bf16)(br * sg + bi * cg);
}

__global__ void lru_prep_C(const float* __restrict__ C_re,
                           const float* __restrict__ C_im,
                           __bf16* __restrict__ Cre_bf,
                           __bf16* __restrict__ Cim_neg_bf) {
    int idx = blockIdx.x * blockDim.x + threadIdx.x;   // H*N
    Cre_bf[idx]     = (__bf16)(C_re[idx]);
    Cim_neg_bf[idx] = (__bf16)(-C_im[idx]);
}

__global__ void lru_conv_x(const float* __restrict__ x, __bf16* __restrict__ x_bf) {
    size_t idx = (size_t)blockIdx.x * blockDim.x + threadIdx.x;  // L*H
    x_bf[idx] = (__bf16)(x[idx]);
}

// ----------------------------- GEMM1: Bu = x @ B'^T ------------------------
// 8 waves / block, each wave computes one 16(L) x 16(N) tile of Bu_re AND Bu_im
// sharing the A (x) fragment. K = H = 512 -> 16 iterations x 2 WMMA.

__global__ void lru_gemm1(const __bf16* __restrict__ x_bf,
                          const __bf16* __restrict__ Bp_re,
                          const __bf16* __restrict__ Bp_im,
                          float* __restrict__ bu_re,
                          float* __restrict__ bu_im) {
    const int lane = threadIdx.x & 31;
    const int wave = blockIdx.x * 8 + (threadIdx.x >> 5);
    const int tile_n = wave & ((LRU_N / 16) - 1);      // 32 tiles over N
    const int tile_l = wave >> 5;                      // 1024 tiles over L
    const int l0 = tile_l * 16;
    const int n0 = tile_n * 16;

    v8f acc_re = {};
    v8f acc_im = {};
#pragma unroll 4
    for (int k0 = 0; k0 < LRU_H; k0 += 32) {
        v16bf a  = load_fragA(x_bf,  LRU_H, l0, k0, lane);
        v16bf br = load_fragB(Bp_re, LRU_H, n0, k0, lane);
        v16bf bi = load_fragB(Bp_im, LRU_H, n0, k0, lane);
        acc_re = __builtin_amdgcn_wmma_f32_16x16x32_bf16(false, a, false, br,
                                                         (short)0, acc_re, false, false);
        acc_im = __builtin_amdgcn_wmma_f32_16x16x32_bf16(false, a, false, bi,
                                                         (short)0, acc_im, false, false);
    }

    const int ncol  = lane & 15;
    const int mbase = (lane >> 4) * 8;
#pragma unroll
    for (int r = 0; r < 8; ++r) {
        size_t o = (size_t)(l0 + mbase + r) * LRU_N + n0 + ncol;
        bu_re[o] = acc_re[r];
        bu_im[o] = acc_im[r];
    }
}

// ------------------------ phase A: chunk-local scan -------------------------
// 256 blocks (one per 64-row chunk) x 512 threads (one per channel).
// In-place: Bu becomes the chunk-local h. Aggregate (chunk final) to agg.

__global__ void lru_scan_local(float* __restrict__ bu_re,
                               float* __restrict__ bu_im,
                               const float* __restrict__ lam_re,
                               const float* __restrict__ lam_im,
                               float* __restrict__ agg_re,
                               float* __restrict__ agg_im) {
    const int n = threadIdx.x;
    const int c = blockIdx.x;
    const float lre = lam_re[n], lim = lam_im[n];
    float hre = 0.f, him = 0.f;
    size_t base = (size_t)c * LRU_CHUNK * LRU_N + n;
    for (int t = 0; t < LRU_CHUNK; ++t) {
        float bre = bu_re[base], bim = bu_im[base];
        float nre = fmaf(lre, hre, fmaf(-lim, him, bre));
        float nim = fmaf(lre, him, fmaf( lim, hre, bim));
        hre = nre; him = nim;
        bu_re[base] = hre;
        bu_im[base] = him;
        base += LRU_N;
    }
    agg_re[c * LRU_N + n] = hre;
    agg_im[c * LRU_N + n] = him;
}

// ------------------- phase B: serial scan over chunk carries ----------------
// One block, 512 threads; carry[c] = state entering chunk c.

__global__ void lru_scan_carry(const float* __restrict__ agg_re,
                               const float* __restrict__ agg_im,
                               const float* __restrict__ lam_re,
                               const float* __restrict__ lam_im,
                               float* __restrict__ carry_re,
                               float* __restrict__ carry_im) {
    const int n = threadIdx.x;
    // lambda^CHUNK by repeated squaring (CHUNK = 64 = 2^6)
    float pr = lam_re[n], pi = lam_im[n];
#pragma unroll
    for (int s = 0; s < 6; ++s) {
        float nr = pr * pr - pi * pi;
        float ni = 2.f * pr * pi;
        pr = nr; pi = ni;
    }
    float Hre = 0.f, Him = 0.f;
    for (int c = 0; c < LRU_NCHUNK; ++c) {
        carry_re[c * LRU_N + n] = Hre;
        carry_im[c * LRU_N + n] = Him;
        float fre = agg_re[c * LRU_N + n];
        float fim = agg_im[c * LRU_N + n];
        float nre = fmaf(pr, Hre, fmaf(-pi, Him, fre));
        float nim = fmaf(pr, Him, fmaf( pi, Hre, fim));
        Hre = nre; Him = nim;
    }
}

// --------------- phase C: apply carries, emit bf16 states -------------------
// h[t] = local[t] + lambda^{t_local+1} * carry(chunk)

__global__ void lru_apply_carry(const float* __restrict__ bu_re,
                                const float* __restrict__ bu_im,
                                const float* __restrict__ carry_re,
                                const float* __restrict__ carry_im,
                                const float* __restrict__ lam_re,
                                const float* __restrict__ lam_im,
                                __bf16* __restrict__ h_re_bf,
                                __bf16* __restrict__ h_im_bf) {
    const int n = threadIdx.x;
    const int c = blockIdx.x;
    const float cre = carry_re[c * LRU_N + n];
    const float cim = carry_im[c * LRU_N + n];
    const float lre = lam_re[n], lim = lam_im[n];
    float pre = lre, pim = lim;     // lambda^1
    size_t base = (size_t)c * LRU_CHUNK * LRU_N + n;
    for (int t = 0; t < LRU_CHUNK; ++t) {
        float hre = bu_re[base] + pre * cre - pim * cim;
        float him = bu_im[base] + pre * cim + pim * cre;
        h_re_bf[base] = (__bf16)hre;
        h_im_bf[base] = (__bf16)him;
        float npr = pre * lre - pim * lim;
        float npi = pre * lim + pim * lre;
        pre = npr; pim = npi;
        base += LRU_N;
    }
}

// -------------------- GEMM2: y = Re(h @ C^T) + D * x ------------------------
// acc = h_re @ C_re^T + h_im @ (-C_im)^T ; fused D*x epilogue.

__global__ void lru_gemm2(const __bf16* __restrict__ h_re_bf,
                          const __bf16* __restrict__ h_im_bf,
                          const __bf16* __restrict__ Cre_bf,
                          const __bf16* __restrict__ Cim_neg_bf,
                          const float* __restrict__ x,
                          const float* __restrict__ D,
                          float* __restrict__ y) {
    const int lane = threadIdx.x & 31;
    const int wave = blockIdx.x * 8 + (threadIdx.x >> 5);
    const int tile_h = wave & ((LRU_H / 16) - 1);      // 32 tiles over H
    const int tile_l = wave >> 5;                      // 1024 tiles over L
    const int l0 = tile_l * 16;
    const int h0 = tile_h * 16;

    v8f acc = {};
#pragma unroll 4
    for (int k0 = 0; k0 < LRU_N; k0 += 32) {
        v16bf are = load_fragA(h_re_bf, LRU_N, l0, k0, lane);
        v16bf aim = load_fragA(h_im_bf, LRU_N, l0, k0, lane);
        v16bf bcr = load_fragB(Cre_bf,     LRU_N, h0, k0, lane);
        v16bf bci = load_fragB(Cim_neg_bf, LRU_N, h0, k0, lane);
        acc = __builtin_amdgcn_wmma_f32_16x16x32_bf16(false, are, false, bcr,
                                                      (short)0, acc, false, false);
        acc = __builtin_amdgcn_wmma_f32_16x16x32_bf16(false, aim, false, bci,
                                                      (short)0, acc, false, false);
    }

    const int ncol  = lane & 15;
    const int mbase = (lane >> 4) * 8;
    const float d = D[h0 + ncol];
#pragma unroll
    for (int r = 0; r < 8; ++r) {
        const int row = l0 + mbase + r;
        const size_t o = (size_t)row * LRU_H + h0 + ncol;
        y[o] = acc[r] + d * x[o];
    }
}

// ------------------------------- launcher ----------------------------------

extern "C" void kernel_launch(void* const* d_in, const int* in_sizes, int n_in,
                              void* d_out, int out_size, void* d_ws, size_t ws_size,
                              hipStream_t stream) {
    (void)in_sizes; (void)n_in; (void)out_size; (void)ws_size;

    const float* x         = (const float*)d_in[0];
    const float* state_re  = (const float*)d_in[1];
    const float* state_im  = (const float*)d_in[2];
    const float* nu_log    = (const float*)d_in[3];
    const float* theta_log = (const float*)d_in[4];
    const float* gamma_log = (const float*)d_in[5];
    const float* B_re      = (const float*)d_in[6];
    const float* B_im      = (const float*)d_in[7];
    const float* C_re      = (const float*)d_in[8];
    const float* C_im      = (const float*)d_in[9];
    const float* Dv        = (const float*)d_in[10];
    float* y = (float*)d_out;

    // workspace carve-up (256B aligned slices)
    char* ws = (char*)d_ws;
    size_t off = 0;
    auto carve = [&](size_t bytes) -> char* {
        char* p = ws + off;
        off += (bytes + 255) & ~(size_t)255;
        return p;
    };
    __bf16* x_bf       = (__bf16*)carve((size_t)LRU_L * LRU_H * 2);
    __bf16* Bp_re      = (__bf16*)carve((size_t)LRU_N * LRU_H * 2);
    __bf16* Bp_im      = (__bf16*)carve((size_t)LRU_N * LRU_H * 2);
    __bf16* Cre_bf     = (__bf16*)carve((size_t)LRU_H * LRU_N * 2);
    __bf16* Cim_neg_bf = (__bf16*)carve((size_t)LRU_H * LRU_N * 2);
    float*  lam_re     = (float*)carve(LRU_N * 4);
    float*  lam_im     = (float*)carve(LRU_N * 4);
    float*  bu_re      = (float*)carve((size_t)LRU_L * LRU_N * 4);
    float*  bu_im      = (float*)carve((size_t)LRU_L * LRU_N * 4);
    float*  agg_re     = (float*)carve((size_t)LRU_NCHUNK * LRU_N * 4);
    float*  agg_im     = (float*)carve((size_t)LRU_NCHUNK * LRU_N * 4);
    float*  carry_re   = (float*)carve((size_t)LRU_NCHUNK * LRU_N * 4);
    float*  carry_im   = (float*)carve((size_t)LRU_NCHUNK * LRU_N * 4);
    __bf16* h_re_bf    = (__bf16*)carve((size_t)LRU_L * LRU_N * 2);
    __bf16* h_im_bf    = (__bf16*)carve((size_t)LRU_L * LRU_N * 2);

    // prep
    lru_prep_lambda<<<1, LRU_N, 0, stream>>>(nu_log, theta_log, state_re, state_im,
                                             lam_re, lam_im);
    lru_prep_B<<<(LRU_N * LRU_H) / 256, 256, 0, stream>>>(B_re, B_im, gamma_log,
                                                          Bp_re, Bp_im);
    lru_prep_C<<<(LRU_H * LRU_N) / 256, 256, 0, stream>>>(C_re, C_im, Cre_bf, Cim_neg_bf);
    lru_conv_x<<<(LRU_L * LRU_H) / 256, 256, 0, stream>>>(x, x_bf);

    // GEMM1: (L/16)*(N/16) = 32768 wave-tiles, 8 waves/block
    lru_gemm1<<<(LRU_L / 16) * (LRU_N / 16) / 8, 256, 0, stream>>>(x_bf, Bp_re, Bp_im,
                                                                   bu_re, bu_im);
    // blocked scan
    lru_scan_local<<<LRU_NCHUNK, LRU_N, 0, stream>>>(bu_re, bu_im, lam_re, lam_im,
                                                     agg_re, agg_im);
    lru_scan_carry<<<1, LRU_N, 0, stream>>>(agg_re, agg_im, lam_re, lam_im,
                                            carry_re, carry_im);
    lru_apply_carry<<<LRU_NCHUNK, LRU_N, 0, stream>>>(bu_re, bu_im, carry_re, carry_im,
                                                      lam_re, lam_im, h_re_bf, h_im_bf);
    // GEMM2 + epilogue
    lru_gemm2<<<(LRU_L / 16) * (LRU_H / 16) / 8, 256, 0, stream>>>(h_re_bf, h_im_bf,
                                                                   Cre_bf, Cim_neg_bf,
                                                                   x, Dv, y);
}
